// MixPalmModel_57930518889107
// MI455X (gfx1250) — matrix operand order repaired and checked by reading, not compiled, not run on hardware
//
#include <hip/hip_runtime.h>
#include <hip/hip_bf16.h>

#define DF 128          // feature width (both layers)
#define WAVES_PER_BLOCK 8

typedef __attribute__((ext_vector_type(16))) __bf16 v16bf;
typedef __attribute__((ext_vector_type(8)))  float  v8f;

static inline int cdiv(long long a, long long b) { return (int)((a + b - 1) / b); }

// ---------------- degree / norm ----------------

__global__ void k_fill1(float* __restrict__ p, int n) {
    int i = blockIdx.x * blockDim.x + threadIdx.x;
    if (i < n) p[i] = 1.0f;                       // self-loop contributes 1 to degree
}

__global__ void k_deg_accum(const long long* __restrict__ ei, float* __restrict__ deg, int nedges) {
    int e = blockIdx.x * blockDim.x + threadIdx.x;
    if (e < nedges) {
        int d = (int)ei[(size_t)nedges + e];      // dst row of edge_index
        atomicAdd(&deg[d], 1.0f);
    }
}

__global__ void k_rsqrt_inplace(float* __restrict__ p, int n) {
    int i = blockIdx.x * blockDim.x + threadIdx.x;
    if (i < n) p[i] = rsqrtf(p[i]);               // deg -> dis = deg^{-1/2}
}

// ---------------- GEMM: C[N,128] = A[N,128] * W[128,128], bf16 WMMA, f32 accum ----------------
// One wave = one 16-row strip, full 128-wide output (8 n-tiles x 4 k-steps).
// W is pre-swizzled into LDS in exact WMMA B-fragment layout:
//   Wl[(nt*4 + ks)*32 + lane] = 16 contiguous bf16 = that lane's B fragment chunk.
// So every B fragment is one aligned 32-byte LDS load (2x ds_load_b128).

__global__ __launch_bounds__(256) void k_gemm_bf16_wmma(const float* __restrict__ A,
                                                        const float* __restrict__ W,
                                                        float* __restrict__ C,
                                                        int nstrips) {
    __shared__ v16bf Wl[8 * 4 * 32];              // 1024 fragments-chunks * 32B = 32 KB

    // Cooperative stage + swizzle: thread builds whole 32-byte fragment chunks.
    for (int f = threadIdx.x; f < 8 * 4 * 32; f += 256) {
        const int lane = f & 31;
        const int ks   = (f >> 5) & 3;
        const int nt   = f >> 7;
        const int hi   = lane >> 4;
        const int n    = nt * 16 + (lane & 15);
        v16bf frag;
#pragma unroll
        for (int v = 0; v < 8; ++v) {
            const int k = ks * 32 + (v >> 2) * 16 + hi * 8 + 2 * (v & 3);   // ISA B 32x16 layout: 2 K per VGPR, table order matching A
            frag[2 * v]     = (__bf16)W[(size_t)k * DF + n];
            frag[2 * v + 1] = (__bf16)W[(size_t)(k + 1) * DF + n];
        }
        Wl[f] = frag;                             // ds_store_b128 x2
    }
    __syncthreads();

    const int lane  = threadIdx.x & 31;
    const int wave  = threadIdx.x >> 5;
    const int strip = blockIdx.x * WAVES_PER_BLOCK + wave;
    if (strip >= nstrips) return;                 // wave-uniform: EXEC stays all-ones for WMMA

    const size_t row0 = (size_t)strip * 16;
    const int m  = lane & 15;                     // A row within tile (ISA 16x32 bf16 A layout)
    const int hi = lane >> 4;                     // lane half selects K sub-range

    // Preload 4 A fragments (K = 0..127). Per lane, each K-step is two contiguous
    // 32-byte chunks: [ks*32 + hi*8, +8) and [ks*32 + 16 + hi*8, +8).
    const float* arow = A + (row0 + (size_t)m) * DF;
    v16bf afrag[4];
#pragma unroll
    for (int ks = 0; ks < 4; ++ks) {
        const float4* p0 = (const float4*)(arow + ks * 32 + hi * 8);
        const float4* p1 = (const float4*)(arow + ks * 32 + 16 + hi * 8);
        const float4 c0a = p0[0], c0b = p0[1];    // K = ks*32 + hi*8 + [0..7]
        const float4 c1a = p1[0], c1b = p1[1];    // K = ks*32 + 16 + hi*8 + [0..7]
        afrag[ks][0]  = (__bf16)c0a.x;  afrag[ks][1]  = (__bf16)c0a.y;
        afrag[ks][2]  = (__bf16)c0a.z;  afrag[ks][3]  = (__bf16)c0a.w;
        afrag[ks][4]  = (__bf16)c0b.x;  afrag[ks][5]  = (__bf16)c0b.y;
        afrag[ks][6]  = (__bf16)c0b.z;  afrag[ks][7]  = (__bf16)c0b.w;
        afrag[ks][8]  = (__bf16)c1a.x;  afrag[ks][9]  = (__bf16)c1a.y;
        afrag[ks][10] = (__bf16)c1a.z;  afrag[ks][11] = (__bf16)c1a.w;
        afrag[ks][12] = (__bf16)c1b.x;  afrag[ks][13] = (__bf16)c1b.y;
        afrag[ks][14] = (__bf16)c1b.z;  afrag[ks][15] = (__bf16)c1b.w;
    }

    const int ncol = lane & 15;                   // B/C column within tile
#pragma unroll
    for (int nt = 0; nt < 8; ++nt) {
        v8f acc = {};
#pragma unroll
        for (int ks = 0; ks < 4; ++ks) {
            const v16bf bfrag = Wl[(nt * 4 + ks) * 32 + lane];   // 2x ds_load_b128
            acc = __builtin_amdgcn_wmma_f32_16x16x32_bf16(
                /*neg_a=*/false, afrag[ks], /*neg_b=*/false, bfrag,
                /*c_mod=*/(short)0, acc, /*reuse_a=*/false, /*reuse_b=*/false);
        }
        // C layout: VGPR v -> row (hi*8 + v), col = ncol
        float* crow = C + (row0 + (size_t)(hi * 8)) * DF + nt * 16 + ncol;
#pragma unroll
        for (int v = 0; v < 8; ++v) crow[(size_t)v * DF] = acc[v];
    }
}

// ---------------- aggregation ----------------
// 32 threads per node/edge, one float4 each (128 floats/row).

__global__ __launch_bounds__(256) void k_selfloop_init(const float* __restrict__ h,
                                                       const float* __restrict__ dis,
                                                       float* __restrict__ out,
                                                       int nnodes) {
    long long t = (long long)blockIdx.x * blockDim.x + threadIdx.x;
    int n = (int)(t >> 5);
    if (n >= nnodes) return;
    int q = (int)(t & 31);
    float dn = dis[n];
    float w  = dn * dn;                           // self-loop norm = 1/deg
    const float4 v = *(const float4*)(h + (size_t)n * DF + q * 4);
    float4 r = make_float4(v.x * w, v.y * w, v.z * w, v.w * w);
    *(float4*)(out + (size_t)n * DF + q * 4) = r;
}

__global__ __launch_bounds__(256) void k_edge_scatter(const long long* __restrict__ ei,
                                                      const float* __restrict__ h,
                                                      const float* __restrict__ dis,
                                                      float* __restrict__ out,
                                                      int nedges) {
    long long t = (long long)blockIdx.x * blockDim.x + threadIdx.x;
    int e = (int)(t >> 5);
    if (e >= nedges) return;
    int q = (int)(t & 31);
    int s = (int)ei[e];
    int d = (int)ei[(size_t)nedges + e];
    float w = dis[s] * dis[d];
    const float4 v = *(const float4*)(h + (size_t)s * DF + q * 4);
    float* o = out + (size_t)d * DF + q * 4;
    atomicAdd(o + 0, v.x * w);                    // global_atomic_add_f32, non-returning
    atomicAdd(o + 1, v.y * w);
    atomicAdd(o + 2, v.z * w);
    atomicAdd(o + 3, v.w * w);
}

__global__ __launch_bounds__(256) void k_bias_relu(float* __restrict__ out,
                                                   const float* __restrict__ b,
                                                   int nnodes) {
    long long t = (long long)blockIdx.x * blockDim.x + threadIdx.x;
    int n = (int)(t >> 5);
    if (n >= nnodes) return;
    int q = (int)(t & 31);
    float4 v = *(float4*)(out + (size_t)n * DF + q * 4);
    const float4 bb = *(const float4*)(b + q * 4);
    v.x = fmaxf(v.x + bb.x, 0.0f);
    v.y = fmaxf(v.y + bb.y, 0.0f);
    v.z = fmaxf(v.z + bb.z, 0.0f);
    v.w = fmaxf(v.w + bb.w, 0.0f);
    *(float4*)(out + (size_t)n * DF + q * 4) = v;
}

// ---------------- launch ----------------

extern "C" void kernel_launch(void* const* d_in, const int* in_sizes, int n_in,
                              void* d_out, int out_size, void* d_ws, size_t ws_size,
                              hipStream_t stream) {
    const float*     x  = (const float*)d_in[0];
    const long long* ei = (const long long*)d_in[1];   // edge_index [2, E] int64
    const float*     W1 = (const float*)d_in[2];
    const float*     b1 = (const float*)d_in[3];
    const float*     W2 = (const float*)d_in[4];
    const float*     b2 = (const float*)d_in[5];
    float*           out = (float*)d_out;

    const int N = in_sizes[0] / DF;                    // 100000
    const int E = in_sizes[1] / 2;                     // 1600000

    // Workspace: dis[N] | h[N*128] | a[N*128]  (~103 MB)
    float* dis = (float*)d_ws;
    float* h   = dis + (((size_t)N + 255) & ~(size_t)255);
    float* a   = h + (size_t)N * DF;

    const int B = 256;
    const int nstrips    = (N + 15) / 16;
    const int gemm_grid  = cdiv(nstrips, WAVES_PER_BLOCK);
    const int node_grid  = cdiv((long long)N * 32, B);
    const int edge_grid  = cdiv((long long)E * 32, B);

    // Degree / normalization
    k_fill1<<<cdiv(N, B), B, 0, stream>>>(dis, N);
    k_deg_accum<<<cdiv(E, B), B, 0, stream>>>(ei, dis, E);
    k_rsqrt_inplace<<<cdiv(N, B), B, 0, stream>>>(dis, N);

    // Layer 1: h = x@W1 ; a = A_hat h ; a = relu(a + b1)
    k_gemm_bf16_wmma<<<gemm_grid, B, 0, stream>>>(x, W1, h, nstrips);
    k_selfloop_init<<<node_grid, B, 0, stream>>>(h, dis, a, N);
    k_edge_scatter<<<edge_grid, B, 0, stream>>>(ei, h, dis, a, E);
    k_bias_relu<<<node_grid, B, 0, stream>>>(a, b1, N);

    // Layer 2: h = a@W2 ; out = A_hat h ; out = relu(out + b2)
    k_gemm_bf16_wmma<<<gemm_grid, B, 0, stream>>>(a, W2, h, nstrips);
    k_selfloop_init<<<node_grid, B, 0, stream>>>(h, dis, out, N);
    k_edge_scatter<<<edge_grid, B, 0, stream>>>(ei, h, dis, out, E);
    k_bias_relu<<<node_grid, B, 0, stream>>>(out, b2, N);
}
